// DeepseekV4Attention_25752623907238
// MI455X (gfx1250) — matrix operand order repaired and checked by reading, not compiled, not run on hardware
//
#include <hip/hip_runtime.h>
#include <hip/hip_bf16.h>

// ---------------------------------------------------------------------------
// Types / helpers
// ---------------------------------------------------------------------------
typedef __attribute__((ext_vector_type(16))) __bf16        bf16x16;
typedef __attribute__((ext_vector_type(8)))  float         f32x8;
typedef __attribute__((ext_vector_type(4)))  unsigned int  u32x4;

struct Pack32 { u32x4 lo, hi; };

__device__ __forceinline__ unsigned short f2bf(float f) {
  unsigned int u = __builtin_bit_cast(unsigned int, f);
  u += 0x7FFFu + ((u >> 16) & 1u);           // round-to-nearest-even
  return (unsigned short)(u >> 16);
}
__device__ __forceinline__ float bf2f(unsigned short h) {
  unsigned int u = ((unsigned int)h) << 16;
  return __builtin_bit_cast(float, u);
}
__device__ __forceinline__ bf16x16 frag_ld(const unsigned short* plo,
                                           const unsigned short* phi) {
  Pack32 t;
  t.lo = *(const u32x4*)plo;
  t.hi = *(const u32x4*)phi;
  return __builtin_bit_cast(bf16x16, t);
}
__device__ __forceinline__ f32x8 wmma_bf16(bf16x16 a, bf16x16 b, f32x8 c) {
  // D = A(16x32) * B(32x16) + C, f32 accumulate
  return __builtin_amdgcn_wmma_f32_16x16x32_bf16(false, a, false, b,
                                                 (short)0, c, false, false);
}
__device__ __forceinline__ f32x8 zero8() {
  f32x8 z = {0.f,0.f,0.f,0.f,0.f,0.f,0.f,0.f};
  return z;
}

// CDNA5 async global->LDS copy (16B per lane), tracked with ASYNCcnt.
// VDST = per-lane LDS byte offset, VADDR = per-lane 64-bit global address.
__device__ __forceinline__ void async_g2l_b128(const void* lds_ptr,
                                               const void* gptr) {
  unsigned lds_off = (unsigned)(size_t)lds_ptr;   // LDS aperture: addr[31:0]
  asm volatile("global_load_async_to_lds_b128 %0, %1, off"
               :
               : "v"(lds_off), "v"((unsigned long long)(size_t)gptr)
               : "memory");
}
__device__ __forceinline__ void wait_async() {
  asm volatile("s_wait_asynccnt 0x0" ::: "memory");
}

// Problem constants
#define Bc     2
#define Sc     2048
#define HIDc   2048
#define Hc     16
#define DHc    192
#define QLORAc 512
#define MROWS  (Bc*Sc)          // 4096

// ---------------------------------------------------------------------------
// f32 -> bf16 conversion
// ---------------------------------------------------------------------------
__global__ void cvt_bf16_kernel(const float* __restrict__ in,
                                unsigned short* __restrict__ out, int n) {
  int i = blockIdx.x * blockDim.x + threadIdx.x;
  if (i < n) out[i] = f2bf(in[i]);
}

// ---------------------------------------------------------------------------
// Generic bf16 GEMM: C[M,N] = A[M,K] * B[K,N]  (A,B bf16 row-major, C f32/bf16)
// Block tile 128x64, K-step 32, 256 threads = 8 waves, wave tile 32x32.
// A tile staged with async global->LDS; B tile transposed through registers.
// ---------------------------------------------------------------------------
template<bool OUT_BF16>
__global__ __launch_bounds__(256)
void gemm_bf16_kernel(const unsigned short* __restrict__ A,
                      const unsigned short* __restrict__ Bm,
                      void* __restrict__ Cv,
                      int M, int N, int K) {
  __shared__ __align__(16) unsigned short sA[128 * 32];   // [row][k]
  __shared__ __align__(16) unsigned short sBt[64 * 32];   // [n][k] (transposed)

  const int tid  = threadIdx.x;
  const int lane = tid & 31;
  const int wave = tid >> 5;
  const int wm   = wave & 3;          // 0..3 -> 32-row slice
  const int wn   = wave >> 2;         // 0..1 -> 32-col slice
  const int r16  = lane & 15;
  const int half = lane >> 4;         // 0/1
  const int mBase = blockIdx.y * 128;
  const int nBase = blockIdx.x * 64;

  f32x8 acc[2][2];
  #pragma unroll
  for (int i = 0; i < 2; ++i)
    #pragma unroll
    for (int j = 0; j < 2; ++j) acc[i][j] = zero8();

  for (int k0 = 0; k0 < K; k0 += 32) {
    __syncthreads();
    // stage A tile 128x32 via async global->LDS (2x b128 per thread)
    #pragma unroll
    for (int c = 0; c < 2; ++c) {
      int idx = tid + c * 256;              // 512 chunks of 8 elems
      int row = idx >> 2, cc = (idx & 3) * 8;
      async_g2l_b128(&sA[row * 32 + cc],
                     &A[(size_t)(mBase + row) * K + k0 + cc]);
    }
    // stage B tile 32x64, transposed into sBt[n][k]
    {
      int krow = tid >> 3, n0 = (tid & 7) * 8;
      u32x4 pk = *(const u32x4*)&Bm[(size_t)(k0 + krow) * N + nBase + n0];
      const unsigned short* e = (const unsigned short*)&pk;
      #pragma unroll
      for (int j = 0; j < 8; ++j) sBt[(n0 + j) * 32 + krow] = e[j];
    }
    if (k0 + 32 < K) {
      __builtin_prefetch(&A[(size_t)(mBase + (tid >> 1)) * K + k0 + 32], 0, 0);
      __builtin_prefetch(&Bm[(size_t)(k0 + 32 + (tid & 31)) * N + nBase], 0, 0);
    }
    wait_async();
    __syncthreads();

    bf16x16 af[2], bfm[2];
    #pragma unroll
    for (int mt = 0; mt < 2; ++mt) {
      const unsigned short* p = &sA[(wm * 32 + mt * 16 + r16) * 32 + half * 8];
      af[mt] = frag_ld(p, p + 16);          // K 0..7(+8h) | 16..23(+8h)
    }
    #pragma unroll
    for (int nt = 0; nt < 2; ++nt) {
      const unsigned short* p = &sBt[(wn * 32 + nt * 16 + r16) * 32 + half * 8];
      bfm[nt] = frag_ld(p, p + 16);         // K 0..7(+8h) | 16..23(+8h)
    }
    #pragma unroll
    for (int mt = 0; mt < 2; ++mt)
      #pragma unroll
      for (int nt = 0; nt < 2; ++nt)
        acc[mt][nt] = wmma_bf16(af[mt], bfm[nt], acc[mt][nt]);
  }

  // store C: row = base + v + 8*(lane>=16), col = base + (lane&15)
  #pragma unroll
  for (int mt = 0; mt < 2; ++mt)
    #pragma unroll
    for (int nt = 0; nt < 2; ++nt)
      #pragma unroll
      for (int vv = 0; vv < 8; ++vv) {
        int row = mBase + wm * 32 + mt * 16 + vv + half * 8;
        int col = nBase + wn * 32 + nt * 16 + r16;
        float val = acc[mt][nt][vv];
        if (OUT_BF16)
          ((unsigned short*)Cv)[(size_t)row * N + col] = f2bf(val);
        else
          ((float*)Cv)[(size_t)row * N + col] = val;
      }
}

// ---------------------------------------------------------------------------
// RMS norm over rows of [MROWS, cols] f32 -> bf16
// ---------------------------------------------------------------------------
__global__ __launch_bounds__(256)
void rmsnorm_kernel(const float* __restrict__ in, const float* __restrict__ w,
                    unsigned short* __restrict__ out, int cols) {
  __shared__ float red[256];
  const int row = blockIdx.x;
  const float* pr = in + (size_t)row * cols;
  float ss = 0.f;
  for (int j = threadIdx.x; j < cols; j += 256) { float t = pr[j]; ss += t * t; }
  red[threadIdx.x] = ss;
  __syncthreads();
  for (int s2 = 128; s2 > 0; s2 >>= 1) {
    if (threadIdx.x < s2) red[threadIdx.x] += red[threadIdx.x + s2];
    __syncthreads();
  }
  float scale = rsqrtf(red[0] / (float)cols + 1e-6f);
  for (int j = threadIdx.x; j < cols; j += 256)
    out[(size_t)row * cols + j] = f2bf(pr[j] * scale * w[j]);
}

// ---------------------------------------------------------------------------
// RoPE (in-place on bf16, layout [B,S,nheads,192], rotary dims 128..191)
// ---------------------------------------------------------------------------
__global__ void rope_kernel(unsigned short* __restrict__ t, int nheads) {
  int idx = blockIdx.x * blockDim.x + threadIdx.x;
  int total = Bc * Sc * nheads * 32;
  if (idx >= total) return;
  int i  = idx & 31;
  int hh = (idx >> 5) % nheads;
  int s  = (idx / (32 * nheads)) % Sc;
  int b  = idx / (32 * nheads * Sc);
  float invf = __powf(10000.0f, -(float)i / 32.0f);
  float ang  = (float)s * invf;
  float cc = __cosf(ang), sn = __sinf(ang);
  unsigned short* base = t + (((size_t)b * Sc + s) * nheads + hh) * DHc + 128;
  float x1 = bf2f(base[i]);
  float x2 = bf2f(base[i + 32]);
  base[i]      = f2bf(x1 * cc - x2 * sn);
  base[i + 32] = f2bf(x2 * cc + x1 * sn);
}

// ---------------------------------------------------------------------------
// Flash attention with sink (MQA): per block = (64 q-rows, head, batch)
// 4 waves. Scores: waves split q-rows; PV: waves split head-dim (48 each).
// Q/K tiles staged with async global->LDS; V staged transposed via registers.
// ---------------------------------------------------------------------------
#define TQ 64
#define TK 32

__global__ __launch_bounds__(128)
void attn_kernel(const unsigned short* __restrict__ q,
                 const unsigned short* __restrict__ k,
                 const unsigned short* __restrict__ v,
                 const float* __restrict__ sink,
                 unsigned short* __restrict__ attn) {
  __shared__ __align__(16) unsigned short sQ[TQ * DHc];   // [r][d]      24.0 KB
  __shared__ __align__(16) unsigned short sKV[TK * DHc];  // K:[t][d] / Vt:[d][t]
  __shared__ __align__(16) float          sS[TQ * TK];    // scores       8.0 KB
  __shared__ __align__(16) unsigned short sP[TQ * TK];    // probs bf16   4.0 KB
  __shared__ float sScale[TQ];

  const int tid  = threadIdx.x;
  const int lane = tid & 31;
  const int wave = tid >> 5;          // 0..3
  const int r16  = lane & 15;
  const int half = lane >> 4;
  const int rowOff = half * 8;        // C-frag row offset for lanes 16..31
  const int qt = blockIdx.x, h = blockIdx.y, b = blockIdx.z;
  const int qBase = qt * TQ;
  const float scaling = 0.07216878364870323f;   // 192^-0.5

  // load Q tile via async global->LDS (12 b128 per thread)
  for (int c = tid; c < TQ * DHc / 8; c += 128) {
    int r = c / (DHc / 8), d8 = c % (DHc / 8);
    async_g2l_b128(&sQ[r * DHc + d8 * 8],
                   &q[(((size_t)b * Sc + qBase + r) * Hc + h) * DHc + d8 * 8]);
  }

  float m_run = 0.f, l_run = 0.f;
  if (tid < TQ) { m_run = sink[h]; l_run = 1.0f; }   // sink == extra logit

  f32x8 oacc[4][3];
  #pragma unroll
  for (int i = 0; i < 4; ++i)
    #pragma unroll
    for (int j = 0; j < 3; ++j) oacc[i][j] = zero8();

  const int ktEnd = 2 * qt + 1;
  for (int kt = 0; kt <= ktEnd; ++kt) {
    __syncthreads();
    // stage K tile [t][d] via async (== transposed-B layout for S = Q*K^T)
    for (int c = tid; c < TK * DHc / 8; c += 128) {
      int t = c / (DHc / 8), d8 = c % (DHc / 8);
      async_g2l_b128(&sKV[t * DHc + d8 * 8],
                     &k[((size_t)b * Sc + kt * TK + t) * DHc + d8 * 8]);
    }
    wait_async();          // covers Q (first iter) + this K tile
    __syncthreads();

    // scores 64x32: wave computes rows [16*wave, 16*wave+16) x all 32 cols
    f32x8 sc[2] = { zero8(), zero8() };
    #pragma unroll
    for (int ks = 0; ks < DHc / 32; ++ks) {
      int k0 = ks * 32;
      const unsigned short* pa = &sQ[(wave * 16 + r16) * DHc + k0 + half * 8];
      bf16x16 aq = frag_ld(pa, pa + 16);
      #pragma unroll
      for (int nt = 0; nt < 2; ++nt) {
        const unsigned short* pb = &sKV[(nt * 16 + r16) * DHc + k0 + half * 8];
        bf16x16 bk = frag_ld(pb, pb + 16);
        sc[nt] = wmma_bf16(aq, bk, sc[nt]);
      }
    }
    #pragma unroll
    for (int nt = 0; nt < 2; ++nt)
      #pragma unroll
      for (int vv = 0; vv < 8; ++vv)
        sS[(wave * 16 + vv + rowOff) * TK + nt * 16 + r16] = sc[nt][vv];
    __syncthreads();

    // stage V transposed [d][t] while row-owners run online softmax
    for (int c = tid; c < TK * DHc / 8; c += 128) {
      int t = c / (DHc / 8), d8 = c % (DHc / 8);
      u32x4 pk = *(const u32x4*)&v[((size_t)b * Sc + kt * TK + t) * DHc + d8 * 8];
      const unsigned short* e = (const unsigned short*)&pk;
      #pragma unroll
      for (int j = 0; j < 8; ++j) sKV[(d8 * 8 + j) * TK + t] = e[j];
    }
    if (tid < TQ) {
      int r = tid, grow = qBase + r;
      float mx = -3.0e38f;
      for (int c2 = 0; c2 < TK; ++c2) {
        int t = kt * TK + c2;
        float sv = sS[r * TK + c2] * scaling;
        if (t <= grow && sv > mx) mx = sv;
      }
      float m_new = fmaxf(m_run, mx);
      float scl = __expf(m_run - m_new);
      float lsum = 0.f;
      for (int c2 = 0; c2 < TK; ++c2) {
        int t = kt * TK + c2;
        float sv = sS[r * TK + c2] * scaling;
        float p = (t <= grow) ? __expf(sv - m_new) : 0.f;
        sP[r * TK + c2] = f2bf(p);
        lsum += p;
      }
      l_run = l_run * scl + lsum;
      m_run = m_new;
      sScale[r] = scl;
    }
    __syncthreads();

    // rescale O accumulators, then O += P * V (wave owns d-range 48*wave..+48)
    #pragma unroll
    for (int mt = 0; mt < 4; ++mt) {
      float rs[8];
      #pragma unroll
      for (int vv = 0; vv < 8; ++vv) rs[vv] = sScale[mt * 16 + vv + rowOff];
      #pragma unroll
      for (int nt = 0; nt < 3; ++nt)
        #pragma unroll
        for (int vv = 0; vv < 8; ++vv) oacc[mt][nt][vv] *= rs[vv];
    }
    bf16x16 ap[4];
    #pragma unroll
    for (int mt = 0; mt < 4; ++mt) {
      const unsigned short* p = &sP[(mt * 16 + r16) * TK + half * 8];
      ap[mt] = frag_ld(p, p + 16);
    }
    #pragma unroll
    for (int nt = 0; nt < 3; ++nt) {
      const unsigned short* pb = &sKV[(wave * 48 + nt * 16 + r16) * TK + half * 8];
      bf16x16 bv = frag_ld(pb, pb + 16);
      #pragma unroll
      for (int mt = 0; mt < 4; ++mt)
        oacc[mt][nt] = wmma_bf16(ap[mt], bv, oacc[mt][nt]);
    }
  }

  __syncthreads();
  if (tid < TQ) sScale[tid] = 1.0f / l_run;
  __syncthreads();

  #pragma unroll
  for (int mt = 0; mt < 4; ++mt)
    #pragma unroll
    for (int nt = 0; nt < 3; ++nt)
      #pragma unroll
      for (int vv = 0; vv < 8; ++vv) {
        int row = mt * 16 + vv + rowOff;
        int d   = wave * 48 + nt * 16 + r16;
        float val = oacc[mt][nt][vv] * sScale[row];
        attn[(((size_t)b * Sc + qBase + row) * Hc + h) * DHc + d] = f2bf(val);
      }
}

// ---------------------------------------------------------------------------
// Host launcher
// ---------------------------------------------------------------------------
extern "C" void kernel_launch(void* const* d_in, const int* in_sizes, int n_in,
                              void* d_out, int out_size, void* d_ws, size_t ws_size,
                              hipStream_t stream) {
  (void)in_sizes; (void)n_in; (void)out_size; (void)ws_size;
  const float* x        = (const float*)d_in[0];
  const float* w_qa     = (const float*)d_in[1];
  const float* q_norm_w = (const float*)d_in[2];
  const float* w_qb     = (const float*)d_in[3];
  const float* w_k      = (const float*)d_in[4];
  const float* w_v      = (const float*)d_in[5];
  const float* w_o      = (const float*)d_in[6];
  const float* sink     = (const float*)d_in[7];
  float* out = (float*)d_out;

  char* ws = (char*)d_ws;
  size_t off = 0;
  auto alloc = [&](size_t bytes) -> char* {
    char* p = ws + off;
    off = (off + bytes + 255) & ~(size_t)255;
    return p;
  };
  unsigned short* x_bf    = (unsigned short*)alloc((size_t)MROWS * HIDc * 2);
  unsigned short* wqa_bf  = (unsigned short*)alloc((size_t)HIDc * QLORAc * 2);
  unsigned short* wqb_bf  = (unsigned short*)alloc((size_t)QLORAc * Hc * DHc * 2);
  unsigned short* wk_bf   = (unsigned short*)alloc((size_t)HIDc * DHc * 2);
  unsigned short* wv_bf   = (unsigned short*)alloc((size_t)HIDc * DHc * 2);
  unsigned short* wo_bf   = (unsigned short*)alloc((size_t)Hc * DHc * HIDc * 2);
  float*          tmp_qa  = (float*)alloc((size_t)MROWS * QLORAc * 4);
  unsigned short* qa_bf   = (unsigned short*)alloc((size_t)MROWS * QLORAc * 2);
  unsigned short* q_bf    = (unsigned short*)alloc((size_t)MROWS * Hc * DHc * 2);
  unsigned short* k_bf    = (unsigned short*)alloc((size_t)MROWS * DHc * 2);
  unsigned short* v_bf    = (unsigned short*)alloc((size_t)MROWS * DHc * 2);
  unsigned short* attn_bf = (unsigned short*)alloc((size_t)MROWS * Hc * DHc * 2);

  auto cvt = [&](const float* src, unsigned short* dst, int n) {
    cvt_bf16_kernel<<<(n + 255) / 256, 256, 0, stream>>>(src, dst, n);
  };
  cvt(x,    x_bf,   MROWS * HIDc);
  cvt(w_qa, wqa_bf, HIDc * QLORAc);
  cvt(w_qb, wqb_bf, QLORAc * Hc * DHc);
  cvt(w_k,  wk_bf,  HIDc * DHc);
  cvt(w_v,  wv_bf,  HIDc * DHc);
  cvt(w_o,  wo_bf,  Hc * DHc * HIDc);

  // q_a = rmsnorm(x @ w_qa)
  gemm_bf16_kernel<false><<<dim3(QLORAc / 64, MROWS / 128), 256, 0, stream>>>(
      x_bf, wqa_bf, tmp_qa, MROWS, QLORAc, HIDc);
  rmsnorm_kernel<<<MROWS, 256, 0, stream>>>(tmp_qa, q_norm_w, qa_bf, QLORAc);

  // q = q_a @ w_qb ; k = x @ w_k ; v = x @ w_v   (bf16 outputs)
  gemm_bf16_kernel<true><<<dim3(Hc * DHc / 64, MROWS / 128), 256, 0, stream>>>(
      qa_bf, wqb_bf, q_bf, MROWS, Hc * DHc, QLORAc);
  gemm_bf16_kernel<true><<<dim3(DHc / 64, MROWS / 128), 256, 0, stream>>>(
      x_bf, wk_bf, k_bf, MROWS, DHc, HIDc);
  gemm_bf16_kernel<true><<<dim3(DHc / 64, MROWS / 128), 256, 0, stream>>>(
      x_bf, wv_bf, v_bf, MROWS, DHc, HIDc);

  // RoPE (in place)
  rope_kernel<<<(Bc * Sc * Hc * 32 + 255) / 256, 256, 0, stream>>>(q_bf, Hc);
  rope_kernel<<<(Bc * Sc * 1 * 32 + 255) / 256, 256, 0, stream>>>(k_bf, 1);

  // flash attention with sink
  attn_kernel<<<dim3(Sc / TQ, Hc, Bc), 128, 0, stream>>>(q_bf, k_bf, v_bf, sink,
                                                         attn_bf);

  // out = attn @ w_o  (f32 output)
  gemm_bf16_kernel<false><<<dim3(HIDc / 64, MROWS / 128), 256, 0, stream>>>(
      attn_bf, wo_bf, out, MROWS, HIDc, Hc * DHc);
}
